// GNNBlock_64252710748259
// MI455X (gfx1250) — matrix operand order repaired and checked by reading, not compiled, not run on hardware
//
#include <hip/hip_runtime.h>

typedef __attribute__((ext_vector_type(16))) _Float16 v16h;
typedef __attribute__((ext_vector_type(8)))  float    v8f;

#define HID 128

// ---------------------------------------------------------------- zero
__global__ void gnn_zero_kernel(float* __restrict__ out, long long n,
                                float* __restrict__ stats, int nstats) {
    long long i = (long long)blockIdx.x * blockDim.x + threadIdx.x;
    long long stride = (long long)gridDim.x * blockDim.x;
    for (; i < n; i += stride) out[i] = 0.0f;
    long long j = (long long)blockIdx.x * blockDim.x + threadIdx.x;
    if (j < nstats) stats[j] = 0.0f;
}

// ---------------------------------------------------------------- edge kernel
// For each 16-edge tile: e = edge_attr(16x32) @ W(32x128) via 8x WMMA f16 (K=32),
// msg = relu(x[src] + e + b), atomic scatter-add into agg[dst].
__global__ __launch_bounds__(256)
void gnn_edge_kernel(const float* __restrict__ x,
                     const long long* __restrict__ ei,   // [2,E] int64
                     const float* __restrict__ ea,       // [E,32]
                     const float* __restrict__ W,        // [32,128]
                     const float* __restrict__ bias,     // [128]
                     float* __restrict__ agg,            // [N,128], pre-zeroed
                     int E) {
    __shared__ _Float16 sW[32 * HID];                    // 8 KB
    for (int i = threadIdx.x; i < 32 * HID; i += blockDim.x)
        sW[i] = (_Float16)W[i];
    __syncthreads();

    const int lane = threadIdx.x & 31;
    const int half = lane >> 4;    // 0: lanes 0-15, 1: lanes 16-31
    const int mrow = lane & 15;

    // Hoist all 8 B fragments (32x16 each) into registers: reused for every edge tile.
    v16h bfrag[8];
    {
        const int kb = half * 16;  // lanes 0-15 hold K=0..15, lanes 16-31 hold K=16..31
        #pragma unroll
        for (int nt = 0; nt < 8; ++nt) {
            const int n = nt * 16 + mrow;
            v16h b;
            #pragma unroll
            for (int i = 0; i < 16; ++i) b[i] = sW[(kb + i) * HID + n];
            bfrag[nt] = b;
        }
    }

    const long long* srcI = ei;
    const long long* dstI = ei + E;

    const int nTiles = E >> 4;
    const int wave   = blockIdx.x * (blockDim.x >> 5) + (threadIdx.x >> 5);
    const int nWaves = gridDim.x * (blockDim.x >> 5);

    for (int t = wave; t < nTiles; t += nWaves) {
        const int e0 = t << 4;

        // Prefetch next tile's streaming operands (lowers to global_prefetch_b8).
        const int tn = t + nWaves;
        if (tn < nTiles) {
            __builtin_prefetch(ea + (size_t)((tn << 4) + mrow) * 32, 0, 1);
            __builtin_prefetch(srcI + (tn << 4), 0, 1);
        }

        // A fragment: 16x32 f16. Row = lane&15; lanes<16: K {0..15}, lanes>=16: {16..31}
        const float* arow = ea + (size_t)(e0 + mrow) * 32;
        const int kb = half * 16;
        v16h a;
        #pragma unroll
        for (int i = 0; i < 8; ++i) {
            a[i]     = (_Float16)arow[kb + i];
            a[8 + i] = (_Float16)arow[kb + 8 + i];
        }

        // Per-lane output rows: m = r + 8*half.  Precompute row pointers once.
        const float* xp[8];
        float*       ap[8];
        #pragma unroll
        for (int r = 0; r < 8; ++r) {
            const int m = r + 8 * half;
            xp[r] = x   + (size_t)((int)srcI[e0 + m]) * HID;
            ap[r] = agg + (size_t)((int)dstI[e0 + m]) * HID;
        }

        #pragma unroll
        for (int nt = 0; nt < 8; ++nt) {
            v8f c = {};
            c = __builtin_amdgcn_wmma_f32_16x16x32_f16(false, a, false, bfrag[nt],
                                                       (short)0, c, false, false);
            const int n    = nt * 16 + mrow;
            const float bc = bias[n];
            // Batch the 8 gathers into one clause (single wait), then 8 atomics.
            float xv[8];
            #pragma unroll
            for (int r = 0; r < 8; ++r) xv[r] = xp[r][n];
            #pragma unroll
            for (int r = 0; r < 8; ++r) {
                float msg = xv[r] + c[r] + bc;
                msg = msg > 0.0f ? msg : 0.0f;
                atomicAdd(ap[r] + n, msg);
            }
        }
    }
}

// ---------------------------------------------------------------- node MLP kernel
// h = agg + x; h1 = relu(h@W1+b1); h2 = h1@W2+b2  (in-place on agg buffer = d_out)
// Weights live in LDS pre-swizzled into WMMA B-fragment order so a fragment load
// is a single 32-byte ds read per lane.  h1 is written to LDS directly in
// A-fragment order, so layer-2 A fragments are also single 32-byte ds reads.
__global__ __launch_bounds__(128)
void gnn_node_kernel(const float* __restrict__ x,
                     const float* __restrict__ W1, const float* __restrict__ b1,
                     const float* __restrict__ W2, const float* __restrict__ b2,
                     float* __restrict__ h,              // in: agg, out: h2 (same rows)
                     float* __restrict__ colsum, float* __restrict__ colsumsq,
                     int N) {
    // Fragment-ordered weights: frag f = kc*8+nt (kc=K/32 chunk, nt=N/16 tile),
    // entry [(f*32 + lane)*16 + elem] = W[k][n] with
    //   k = kc*32 + (lane>>4)*8 + (elem&7) + (elem>>3)*16 ,  n = nt*16 + (lane&15)
    __shared__ __align__(32) _Float16 sWf[32 * 32 * 16];   // 32 KB
    // Per-wave h1 scratch in A-fragment order: [(kc*32 + lane)*16 + elem]
    __shared__ __align__(32) _Float16 sHf[4][4 * 32 * 16]; // 16 KB
    __shared__ float sSum[HID], sSq[HID];

    const int lane = threadIdx.x & 31;
    const int wv   = threadIdx.x >> 5;     // 0..3
    const int half = lane >> 4;
    const int mrow = lane & 15;

    for (int i = threadIdx.x; i < HID; i += blockDim.x) { sSum[i] = 0.0f; sSq[i] = 0.0f; }

    const v16h* wf = (const v16h*)sWf;

    const int nTiles = N >> 4;
    for (int base = blockIdx.x * 4; base < nTiles; base += gridDim.x * 4) {
        const int t = base + wv;

        // ---- phase 1: W1 (fragment-swizzled) in LDS, compute h1 into sHf ----
        __syncthreads();
        for (int i = threadIdx.x; i < 32 * 32 * 16; i += blockDim.x) {
            const int elem = i & 15;
            const int ln   = (i >> 4) & 31;
            const int fidx = i >> 9;            // kc*8 + nt
            const int k    = (fidx >> 3) * 32 + ((ln >> 4) * 8) + (elem & 7) + ((elem >> 3) * 16);
            const int n    = (fidx & 7) * 16 + (ln & 15);
            sWf[i] = (_Float16)W1[k * HID + n];
        }
        __syncthreads();

        if (t < nTiles) {
            const int row = t * 16 + mrow;
            const float* hr = h + (size_t)row * HID;
            const float* xr = x + (size_t)row * HID;
            v16h afr[4];
            #pragma unroll
            for (int kc = 0; kc < 4; ++kc) {
                const int kb = kc * 32 + half * 8;
                v16h av;
                #pragma unroll
                for (int i = 0; i < 8; ++i) {
                    av[i]     = (_Float16)(hr[kb + i]      + xr[kb + i]);
                    av[8 + i] = (_Float16)(hr[kb + 16 + i] + xr[kb + 16 + i]);
                }
                afr[kc] = av;
            }
            #pragma unroll
            for (int nt = 0; nt < 8; ++nt) {
                const int n = nt * 16 + mrow;
                v8f c = {};
                #pragma unroll
                for (int kc = 0; kc < 4; ++kc)
                    c = __builtin_amdgcn_wmma_f32_16x16x32_f16(false, afr[kc], false,
                                                               wf[(kc * 8 + nt) * 32 + lane],
                                                               (short)0, c, false, false);
                const float bb = b1[n];
                // Scatter h1 into A-fragment order for layer 2:
                // value (m=r+8*half, col=n) -> frag kc2=n>>5, lane' = hi*16 + m, elem'
                const int kc2     = n >> 5;
                const int within  = n & 31;
                const int lane_hi = (within >> 3) & 1;
                const int elem2   = (within & 7) + ((within & 16) ? 8 : 0);
                #pragma unroll
                for (int r = 0; r < 8; ++r) {
                    float v = c[r] + bb;
                    v = v > 0.0f ? v : 0.0f;
                    sHf[wv][(kc2 * 32 + lane_hi * 16 + r + 8 * half) * 16 + elem2] = (_Float16)v;
                }
            }
        }
        __syncthreads();

        // ---- phase 2: W2 (fragment-swizzled) in LDS, compute h2, stats ----
        for (int i = threadIdx.x; i < 32 * 32 * 16; i += blockDim.x) {
            const int elem = i & 15;
            const int ln   = (i >> 4) & 31;
            const int fidx = i >> 9;
            const int k    = (fidx >> 3) * 32 + ((ln >> 4) * 8) + (elem & 7) + ((elem >> 3) * 16);
            const int n    = (fidx & 7) * 16 + (ln & 15);
            sWf[i] = (_Float16)W2[k * HID + n];
        }
        __syncthreads();

        if (t < nTiles) {
            const v16h* hf = (const v16h*)sHf[wv];
            v16h afr[4];
            #pragma unroll
            for (int kc = 0; kc < 4; ++kc) afr[kc] = hf[kc * 32 + lane];

            #pragma unroll
            for (int nt = 0; nt < 8; ++nt) {
                const int n = nt * 16 + mrow;
                v8f c = {};
                #pragma unroll
                for (int kc = 0; kc < 4; ++kc)
                    c = __builtin_amdgcn_wmma_f32_16x16x32_f16(false, afr[kc], false,
                                                               wf[(kc * 8 + nt) * 32 + lane],
                                                               (short)0, c, false, false);
                const float bb = b2[n];
                float ps = 0.0f, pq = 0.0f;
                #pragma unroll
                for (int r = 0; r < 8; ++r) {
                    const int row = t * 16 + r + 8 * half;
                    float v = c[r] + bb;
                    h[(size_t)row * HID + n] = v;
                    ps += v; pq += v * v;
                }
                atomicAdd(&sSum[n], ps);
                atomicAdd(&sSq[n], pq);
            }
        }
    }
    __syncthreads();
    for (int i = threadIdx.x; i < HID; i += blockDim.x) {
        atomicAdd(&colsum[i], sSum[i]);
        atomicAdd(&colsumsq[i], sSq[i]);
    }
}

// ---------------------------------------------------------------- GraphNorm stats
// var = E[h^2] - (2s - s^2) * mean^2 ; fold into per-column affine: out = relu(h*A + B)
__global__ void gnn_stats_kernel(const float* __restrict__ colsum,
                                 const float* __restrict__ colsumsq,
                                 const float* __restrict__ gw,
                                 const float* __restrict__ gb,
                                 const float* __restrict__ gs,
                                 float* __restrict__ A, float* __restrict__ B, int N) {
    const int i = threadIdx.x;
    if (i < HID) {
        const float invN = 1.0f / (float)N;
        const float m = colsum[i] * invN;
        const float q = colsumsq[i] * invN;
        const float s = gs[i];
        const float var = q - (2.0f * s - s * s) * m * m;
        const float a = gw[i] * rsqrtf(var + 1e-5f);
        A[i] = a;
        B[i] = gb[i] - a * s * m;
    }
}

// ---------------------------------------------------------------- normalize + relu
__global__ void gnn_norm_kernel(float* __restrict__ h,
                                const float* __restrict__ A,
                                const float* __restrict__ B, long long total) {
    long long i = (long long)blockIdx.x * blockDim.x + threadIdx.x;
    long long stride = (long long)gridDim.x * blockDim.x;
    for (; i < total; i += stride) {
        const int col = (int)(i & (HID - 1));
        float v = h[i] * A[col] + B[col];
        h[i] = v > 0.0f ? v : 0.0f;
    }
}

// ---------------------------------------------------------------- launch
extern "C" void kernel_launch(void* const* d_in, const int* in_sizes, int n_in,
                              void* d_out, int out_size, void* d_ws, size_t ws_size,
                              hipStream_t stream) {
    const float*      x  = (const float*)d_in[0];
    const long long*  ei = (const long long*)d_in[1];   // jnp.int64 edge_index [2,E]
    const float*      ea = (const float*)d_in[2];
    const float*      eW = (const float*)d_in[3];
    const float*      eb = (const float*)d_in[4];
    const float*      W1 = (const float*)d_in[5];
    const float*      b1 = (const float*)d_in[6];
    const float*      W2 = (const float*)d_in[7];
    const float*      b2 = (const float*)d_in[8];
    const float*      gw = (const float*)d_in[9];
    const float*      gb = (const float*)d_in[10];
    const float*      gs = (const float*)d_in[11];
    float* out = (float*)d_out;

    const int N = in_sizes[0] / HID;
    const int E = in_sizes[2] / 32;

    float* colsum   = (float*)d_ws;        // [128]
    float* colsumsq = colsum + HID;        // [128]
    float* scA      = colsum + 2 * HID;    // [128]
    float* scB      = colsum + 3 * HID;    // [128]

    const long long total = (long long)N * HID;

    gnn_zero_kernel<<<2048, 256, 0, stream>>>(out, total, colsum, 2 * HID);
    gnn_edge_kernel<<<2048, 256, 0, stream>>>(x, ei, ea, eW, eb, out, E);
    gnn_node_kernel<<<1024, 128, 0, stream>>>(x, W1, b1, W2, b2, out, colsum, colsumsq, N);
    gnn_stats_kernel<<<1, 128, 0, stream>>>(colsum, colsumsq, gw, gb, gs, scA, scB, N);
    gnn_norm_kernel<<<2048, 256, 0, stream>>>(out, scA, scB, total);
}
